// PMFLoss_7301444403233
// MI455X (gfx1250) — matrix operand order repaired and checked by reading, not compiled, not run on hardware
//
#include <hip/hip_runtime.h>
#include <hip/hip_bf16.h>
#include <math.h>

typedef __attribute__((ext_vector_type(16))) _Float16 v16h;
typedef __attribute__((ext_vector_type(8)))  float    v8f;

#define EMB        32
#define NBLOCKS    1024
#define NTHREADS   256
#define WPB        (NTHREADS / 32)   // waves per block
#define LAM_U      0.1f

// Stage 1: each wave handles groups of 16 interactions.
//   - gather 16 u-rows into an A fragment (16x32 f16), 16 v-rows into B (32x16 f16)
//   - one v_wmma_f32_16x16x32_f16 computes all 16 dot products on its diagonal
//   - Su^2 / Sv^2 accumulated from the f32 values (exact lane partition of rows)
//   - stable BCE term on the 16 diagonal-owning lanes
// Deterministic: fixed shuffle tree + per-block partial slots (no float atomics).
__global__ void __launch_bounds__(NTHREADS)
pmf_partial_kernel(const int*   __restrict__ user,
                   const int*   __restrict__ item,
                   const float* __restrict__ ratings,
                   const float* __restrict__ Ue,
                   const float* __restrict__ Ve,
                   float*       __restrict__ partials,  // [gridDim.x * 3]
                   int B)
{
    const int lane = threadIdx.x & 31;   // wave32
    const int wave = threadIdx.x >> 5;
    const int half = lane >> 4;          // 0: lanes 0-15, 1: lanes 16-31
    const int m    = lane & 15;          // row (A) / column (B) slot in group

    const int nGroups    = (B + 15) >> 4;
    const int wavesTotal = (int)gridDim.x * WPB;
    const int waveId     = (int)blockIdx.x * WPB + wave;

    float bce = 0.0f, usq = 0.0f, vsq = 0.0f;

    for (int g = waveId; g < nGroups; g += wavesTotal) {
        const int  base     = g << 4;
        const int  ii       = base + m;
        const bool rowValid = (ii < B);
        const int  iSafe    = rowValid ? ii : 0;

        // prefetch next group's index/rating cachelines (global_prefetch_b8)
        const int gn = g + wavesTotal;
        if (gn < nGroups) {
            const int nb = gn << 4;
            __builtin_prefetch(user    + nb, 0, 0);
            __builtin_prefetch(item    + nb, 0, 0);
            __builtin_prefetch(ratings + nb, 0, 0);
        }

        const float* urow = Ue + (size_t)user[iSafe] * EMB;
        const float* vrow = Ve + (size_t)item[iSafe] * EMB;

        // ---- A fragment (u): this lane owns K {k0..k0+7} U {k0+16..k0+23} of row m
        const int k0 = half * 8;
        float4 a0 = *(const float4*)(urow + k0);
        float4 a1 = *(const float4*)(urow + k0 + 4);
        float4 a2 = *(const float4*)(urow + k0 + 16);
        float4 a3 = *(const float4*)(urow + k0 + 20);

        // ---- B fragment (v): this lane owns K {kb..kb+7} U {kb+16..kb+23} of column m
        const int kb = half * 8;
        float4 b0 = *(const float4*)(vrow + kb);
        float4 b1 = *(const float4*)(vrow + kb + 4);
        float4 b2 = *(const float4*)(vrow + kb + 16);
        float4 b3 = *(const float4*)(vrow + kb + 20);

        if (rowValid) {
            usq += a0.x*a0.x + a0.y*a0.y + a0.z*a0.z + a0.w*a0.w
                 + a1.x*a1.x + a1.y*a1.y + a1.z*a1.z + a1.w*a1.w
                 + a2.x*a2.x + a2.y*a2.y + a2.z*a2.z + a2.w*a2.w
                 + a3.x*a3.x + a3.y*a3.y + a3.z*a3.z + a3.w*a3.w;
            vsq += b0.x*b0.x + b0.y*b0.y + b0.z*b0.z + b0.w*b0.w
                 + b1.x*b1.x + b1.y*b1.y + b1.z*b1.z + b1.w*b1.w
                 + b2.x*b2.x + b2.y*b2.y + b2.z*b2.z + b2.w*b2.w
                 + b3.x*b3.x + b3.y*b3.y + b3.z*b3.z + b3.w*b3.w;
        }

        // pack f32 -> f16 fragments (ISA 16-bit A/B wave32 layouts)
        v16h Af, Bf;
        Af[0]=(_Float16)a0.x; Af[1]=(_Float16)a0.y; Af[2] =(_Float16)a0.z; Af[3] =(_Float16)a0.w;
        Af[4]=(_Float16)a1.x; Af[5]=(_Float16)a1.y; Af[6] =(_Float16)a1.z; Af[7] =(_Float16)a1.w;
        Af[8]=(_Float16)a2.x; Af[9]=(_Float16)a2.y; Af[10]=(_Float16)a2.z; Af[11]=(_Float16)a2.w;
        Af[12]=(_Float16)a3.x;Af[13]=(_Float16)a3.y;Af[14]=(_Float16)a3.z; Af[15]=(_Float16)a3.w;
        Bf[0]=(_Float16)b0.x; Bf[1]=(_Float16)b0.y; Bf[2] =(_Float16)b0.z; Bf[3] =(_Float16)b0.w;
        Bf[4]=(_Float16)b1.x; Bf[5]=(_Float16)b1.y; Bf[6] =(_Float16)b1.z; Bf[7] =(_Float16)b1.w;
        Bf[8]=(_Float16)b2.x; Bf[9]=(_Float16)b2.y; Bf[10]=(_Float16)b2.z; Bf[11]=(_Float16)b2.w;
        Bf[12]=(_Float16)b3.x;Bf[13]=(_Float16)b3.y;Bf[14]=(_Float16)b3.z; Bf[15]=(_Float16)b3.w;

        // 16 logits on the diagonal of D (EXEC is all-1s here: no divergence)
        v8f c = {};
        c = __builtin_amdgcn_wmma_f32_16x16x32_f16(
                /*neg_a=*/false, Af, /*neg_b=*/false, Bf,
                /*c_mod=*/(short)0, c, /*reuse_a=*/false, /*reuse_b=*/false);

        // Diagonal: lanes 0-7 -> D[l][l] in c[l]; lanes 24-31 -> D[l-16][l-16] in c[l-24]
        const bool diagLane = (lane < 8) || (lane >= 24);
        if (diagLane) {
            const int r  = (lane < 8) ? lane : (lane - 24);  // element index 0..7
            const int li = (lane < 8) ? lane : (lane - 16);  // interaction slot 0..15
            float x = c[0];
            x = (r == 1) ? (float)c[1] : x;
            x = (r == 2) ? (float)c[2] : x;
            x = (r == 3) ? (float)c[3] : x;
            x = (r == 4) ? (float)c[4] : x;
            x = (r == 5) ? (float)c[5] : x;
            x = (r == 6) ? (float)c[6] : x;
            x = (r == 7) ? (float)c[7] : x;
            const int oi = base + li;
            if (oi < B) {
                const float y  = ratings[oi];
                const float t  = __expf(-fabsf(x));
                bce += fmaxf(x, 0.0f) + __logf(1.0f + t) - x * y;
            }
        }
    }

    // deterministic wave tree reduction
    for (int off = 16; off > 0; off >>= 1) {
        bce += __shfl_xor(bce, off, 32);
        usq += __shfl_xor(usq, off, 32);
        vsq += __shfl_xor(vsq, off, 32);
    }

    __shared__ float sB[WPB], sU[WPB], sV[WPB];
    if (lane == 0) { sB[wave] = bce; sU[wave] = usq; sV[wave] = vsq; }
    __syncthreads();
    if (threadIdx.x == 0) {
        float tb = 0.0f, tu = 0.0f, tv = 0.0f;
        for (int w = 0; w < WPB; ++w) { tb += sB[w]; tu += sU[w]; tv += sV[w]; }
        partials[(size_t)blockIdx.x * 3 + 0] = tb;
        partials[(size_t)blockIdx.x * 3 + 1] = tu;
        partials[(size_t)blockIdx.x * 3 + 2] = tv;
    }
}

// Stage 2: fixed-order reduction over block partials + final scalar combine.
__global__ void __launch_bounds__(NTHREADS)
pmf_final_kernel(const float* __restrict__ partials, float* __restrict__ out, int nParts)
{
    __shared__ float sB[NTHREADS], sU[NTHREADS], sV[NTHREADS];
    float tb = 0.0f, tu = 0.0f, tv = 0.0f;
    for (int i = threadIdx.x; i < nParts; i += NTHREADS) {
        tb += partials[(size_t)i * 3 + 0];
        tu += partials[(size_t)i * 3 + 1];
        tv += partials[(size_t)i * 3 + 2];
    }
    sB[threadIdx.x] = tb; sU[threadIdx.x] = tu; sV[threadIdx.x] = tv;
    __syncthreads();
    for (int s = NTHREADS / 2; s > 0; s >>= 1) {
        if (threadIdx.x < s) {
            sB[threadIdx.x] += sB[threadIdx.x + s];
            sU[threadIdx.x] += sU[threadIdx.x + s];
            sV[threadIdx.x] += sV[threadIdx.x + s];
        }
        __syncthreads();
    }
    if (threadIdx.x == 0)
        out[0] = sB[0] + LAM_U * sqrtf(sU[0]) + LAM_U * sqrtf(sV[0]);
}

extern "C" void kernel_launch(void* const* d_in, const int* in_sizes, int n_in,
                              void* d_out, int out_size, void* d_ws, size_t ws_size,
                              hipStream_t stream) {
    const int*   user    = (const int*)  d_in[0];
    const int*   item    = (const int*)  d_in[1];
    const float* ratings = (const float*)d_in[2];
    const float* Ue      = (const float*)d_in[3];
    const float* Ve      = (const float*)d_in[4];
    float*       out     = (float*)d_out;
    float*       parts   = (float*)d_ws;          // NBLOCKS*3 floats (12 KB)
    const int    B       = in_sizes[0];

    pmf_partial_kernel<<<NBLOCKS, NTHREADS, 0, stream>>>(user, item, ratings, Ue, Ve, parts, B);
    pmf_final_kernel<<<1, NTHREADS, 0, stream>>>(parts, out, NBLOCKS);
}
